// Joint_14001593385645
// MI455X (gfx1250) — compile-verified, matrix operand order checked by reading, not run on hardware
//
#include <hip/hip_runtime.h>

typedef __attribute__((ext_vector_type(16))) _Float16 v16h;
typedef __attribute__((ext_vector_type(8)))  _Float16 v8h;
typedef __attribute__((ext_vector_type(4)))  _Float16 v4h;
typedef __attribute__((ext_vector_type(8)))  float    v8f;

#define D_MODEL  512
#define D_HIDDEN 1024
#define D_OUT    128
#define BATCH    8
#define T_ENC    256
#define U_DEC    64
#define SH_STRIDE 1032   // 1024 + 8 halves pad -> 4-dword bank stagger per row
#define SA_STRIDE 520    // 512 + 8 halves pad

union FragH { v16h v; v8h h[2]; };

__device__ __forceinline__ float fast_tanh(float x) {
  // tanh(x) = 1 - 2/(e^{2x}+1); exp via v_exp_f32 (TRANS, co-executes with WMMA)
  float e = __builtin_amdgcn_exp2f(x * 2.8853900817779268f); // 2*log2(e)
  return 1.0f - 2.0f * __builtin_amdgcn_rcpf(e + 1.0f);
}

// ---------------------------------------------------------------------------
// prep: W1t[h][d] = f16(W1[d][h])  (1024x1024), W2t[o][h] = f16(W2[h][o]) (128x1024)
// ---------------------------------------------------------------------------
__global__ __launch_bounds__(256) void prep_kernel(const float* __restrict__ W1,
                                                   const float* __restrict__ W2,
                                                   _Float16* __restrict__ W1t,
                                                   _Float16* __restrict__ W2t) {
  int idx = blockIdx.x * 256 + threadIdx.x;
  if (idx < 1024 * 1024) {
    int h = idx >> 10, d = idx & 1023;
    W1t[idx] = (_Float16)W1[d * 1024 + h];
  } else {
    int i = idx - 1024 * 1024;            // o*1024 + h
    int o = i >> 10, h = i & 1023;
    W2t[i] = (_Float16)W2[h * 128 + o];
  }
}

// ---------------------------------------------------------------------------
// proj: Out[M,1024] = A[M,512] @ Bt^T (+bias).
// Grid: (M/64, 8). Block = 64 M-rows x 128 N-cols; A-tile 64x512 f16 in LDS.
// Each wave: one 16-wide N-tile, 4 M-tile accumulators, single pipelined
// k-loop (no outer loop -> nothing for LICM to hoist/spill).
// ---------------------------------------------------------------------------
__global__ __launch_bounds__(256) void proj_kernel(const float* __restrict__ A,
                                                   const _Float16* __restrict__ Bt,
                                                   const float* __restrict__ bias,
                                                   float* __restrict__ Out) {
  __shared__ _Float16 sA[64 * SA_STRIDE];
  const int tid  = threadIdx.x;
  const int lane = tid & 31;
  const int wave = tid >> 5;
  const int row0 = blockIdx.x * 64;

  // fill: 64 rows x 512 cols, float4 loads, 8B LDS stores
  #pragma unroll 4
  for (int j = 0; j < 32; ++j) {
    int idx = j * 256 + tid;              // float4 index
    int r = idx >> 7, c4 = idx & 127;     // 128 float4 per row
    float4 v = ((const float4*)(A + (size_t)(row0 + r) * D_MODEL))[c4];
    v4h hv;
    hv[0] = (_Float16)v.x; hv[1] = (_Float16)v.y;
    hv[2] = (_Float16)v.z; hv[3] = (_Float16)v.w;
    *(v4h*)&sA[r * SA_STRIDE + c4 * 4] = hv;
  }
  __syncthreads();

  const int lr   = lane & 15;
  const int khi  = (lane & 16) ? 8 : 0;
  const int mofs = (lane & 16) ? 8 : 0;
  const int n0   = (blockIdx.y * 8 + wave) * 16;
  const _Float16* ap = &sA[lr * SA_STRIDE + khi];
  const _Float16* bp = &Bt[(n0 + lr) * 1024 + khi];

  v8f acc0 = {}, acc1 = {}, acc2 = {}, acc3 = {};
  FragH bc, a0, a1, a2, a3;
  bc.h[0] = *(const v8h*)bp;
  bc.h[1] = *(const v8h*)(bp + 16);
  a0.h[0] = *(const v8h*)(ap + 0 * 16 * SA_STRIDE);
  a0.h[1] = *(const v8h*)(ap + 0 * 16 * SA_STRIDE + 16);
  a1.h[0] = *(const v8h*)(ap + 1 * 16 * SA_STRIDE);
  a1.h[1] = *(const v8h*)(ap + 1 * 16 * SA_STRIDE + 16);
  a2.h[0] = *(const v8h*)(ap + 2 * 16 * SA_STRIDE);
  a2.h[1] = *(const v8h*)(ap + 2 * 16 * SA_STRIDE + 16);
  a3.h[0] = *(const v8h*)(ap + 3 * 16 * SA_STRIDE);
  a3.h[1] = *(const v8h*)(ap + 3 * 16 * SA_STRIDE + 16);

  #pragma unroll
  for (int kk = 0; kk < D_MODEL / 32 - 1; ++kk) {
    const int kb = (kk + 1) * 32;
    FragH bn, n0f, n1f, n2f, n3f;
    bn.h[0]  = *(const v8h*)(bp + kb);
    bn.h[1]  = *(const v8h*)(bp + kb + 16);
    n0f.h[0] = *(const v8h*)(ap + kb + 0 * 16 * SA_STRIDE);
    n0f.h[1] = *(const v8h*)(ap + kb + 0 * 16 * SA_STRIDE + 16);
    n1f.h[0] = *(const v8h*)(ap + kb + 1 * 16 * SA_STRIDE);
    n1f.h[1] = *(const v8h*)(ap + kb + 1 * 16 * SA_STRIDE + 16);
    n2f.h[0] = *(const v8h*)(ap + kb + 2 * 16 * SA_STRIDE);
    n2f.h[1] = *(const v8h*)(ap + kb + 2 * 16 * SA_STRIDE + 16);
    n3f.h[0] = *(const v8h*)(ap + kb + 3 * 16 * SA_STRIDE);
    n3f.h[1] = *(const v8h*)(ap + kb + 3 * 16 * SA_STRIDE + 16);

    acc0 = __builtin_amdgcn_wmma_f32_16x16x32_f16(false, a0.v, false, bc.v,
                                                  (short)0, acc0, false, false);
    acc1 = __builtin_amdgcn_wmma_f32_16x16x32_f16(false, a1.v, false, bc.v,
                                                  (short)0, acc1, false, false);
    acc2 = __builtin_amdgcn_wmma_f32_16x16x32_f16(false, a2.v, false, bc.v,
                                                  (short)0, acc2, false, false);
    acc3 = __builtin_amdgcn_wmma_f32_16x16x32_f16(false, a3.v, false, bc.v,
                                                  (short)0, acc3, false, false);
    bc = bn; a0 = n0f; a1 = n1f; a2 = n2f; a3 = n3f;
  }
  acc0 = __builtin_amdgcn_wmma_f32_16x16x32_f16(false, a0.v, false, bc.v,
                                                (short)0, acc0, false, false);
  acc1 = __builtin_amdgcn_wmma_f32_16x16x32_f16(false, a1.v, false, bc.v,
                                                (short)0, acc1, false, false);
  acc2 = __builtin_amdgcn_wmma_f32_16x16x32_f16(false, a2.v, false, bc.v,
                                                (short)0, acc2, false, false);
  acc3 = __builtin_amdgcn_wmma_f32_16x16x32_f16(false, a3.v, false, bc.v,
                                                (short)0, acc3, false, false);

  const float bv = bias ? bias[n0 + lr] : 0.0f;
  #pragma unroll
  for (int r = 0; r < 8; ++r) {
    Out[(size_t)(row0 +  0 + r + mofs) * D_HIDDEN + n0 + lr] = acc0[r] + bv;
    Out[(size_t)(row0 + 16 + r + mofs) * D_HIDDEN + n0 + lr] = acc1[r] + bv;
    Out[(size_t)(row0 + 32 + r + mofs) * D_HIDDEN + n0 + lr] = acc2[r] + bv;
    Out[(size_t)(row0 + 48 + r + mofs) * D_HIDDEN + n0 + lr] = acc3[r] + bv;
  }
}

// ---------------------------------------------------------------------------
// joint: block = one (b,t), all 64 u rows. tanh tile 64x1024 f16 in LDS
// (~129KB), then each wave owns one 16-wide N-tile with 4 M-tile accumulators:
// per k-step 1 B-frag (global, 4x reused) + 4 A-frags (LDS) -> 4 WMMA,
// software-pipelined one step ahead.
// ---------------------------------------------------------------------------
__global__ __launch_bounds__(256) void joint_kernel(const float* __restrict__ enc_proj,
                                                    const float* __restrict__ dec_projb1,
                                                    const _Float16* __restrict__ W2t,
                                                    const float* __restrict__ b2,
                                                    float* __restrict__ Out) {
  __shared__ _Float16 sH[U_DEC * SH_STRIDE];
  const int tid  = threadIdx.x;
  const int lane = tid & 31;
  const int wave = tid >> 5;

  const int bt = blockIdx.x;              // b*T + t
  const int b  = bt >> 8;                 // T = 256

  const float* ep = &enc_proj[(size_t)bt * D_HIDDEN];
  const float* dp = &dec_projb1[(size_t)b * U_DEC * D_HIDDEN];

  // ---- activation tile: 64 rows x 1024 cols, float4 loads, 8B LDS stores ----
  #pragma unroll 4
  for (int j = 0; j < 64; ++j) {
    int idx = j * 256 + tid;              // float4 index
    int r = idx >> 8, c4 = idx & 255;     // 256 float4 per row
    float4 e = ((const float4*)ep)[c4];
    float4 d = ((const float4*)(dp + (size_t)r * D_HIDDEN))[c4];
    v4h hv;
    hv[0] = (_Float16)fast_tanh(e.x + d.x);
    hv[1] = (_Float16)fast_tanh(e.y + d.y);
    hv[2] = (_Float16)fast_tanh(e.z + d.z);
    hv[3] = (_Float16)fast_tanh(e.w + d.w);
    *(v4h*)&sH[r * SH_STRIDE + c4 * 4] = hv;
  }
  __syncthreads();

  // ---- WMMA phase ----
  const int lr   = lane & 15;
  const int khi  = (lane & 16) ? 8 : 0;
  const int mofs = (lane & 16) ? 8 : 0;
  const int n0   = wave * 16;
  const _Float16* bp = &W2t[(n0 + lr) * 1024 + khi];
  const _Float16* ap = &sH[lr * SH_STRIDE + khi];

  v8f acc0 = {}, acc1 = {}, acc2 = {}, acc3 = {};
  FragH bc, a0, a1, a2, a3;

  bc.h[0] = *(const v8h*)bp;
  bc.h[1] = *(const v8h*)(bp + 16);
  a0.h[0] = *(const v8h*)(ap + 0 * 16 * SH_STRIDE);
  a0.h[1] = *(const v8h*)(ap + 0 * 16 * SH_STRIDE + 16);
  a1.h[0] = *(const v8h*)(ap + 1 * 16 * SH_STRIDE);
  a1.h[1] = *(const v8h*)(ap + 1 * 16 * SH_STRIDE + 16);
  a2.h[0] = *(const v8h*)(ap + 2 * 16 * SH_STRIDE);
  a2.h[1] = *(const v8h*)(ap + 2 * 16 * SH_STRIDE + 16);
  a3.h[0] = *(const v8h*)(ap + 3 * 16 * SH_STRIDE);
  a3.h[1] = *(const v8h*)(ap + 3 * 16 * SH_STRIDE + 16);

  #pragma unroll
  for (int kk = 0; kk < D_HIDDEN / 32 - 1; ++kk) {
    const int kb = (kk + 1) * 32;
    FragH bn, n0f, n1f, n2f, n3f;
    bn.h[0]  = *(const v8h*)(bp + kb);
    bn.h[1]  = *(const v8h*)(bp + kb + 16);
    n0f.h[0] = *(const v8h*)(ap + kb + 0 * 16 * SH_STRIDE);
    n0f.h[1] = *(const v8h*)(ap + kb + 0 * 16 * SH_STRIDE + 16);
    n1f.h[0] = *(const v8h*)(ap + kb + 1 * 16 * SH_STRIDE);
    n1f.h[1] = *(const v8h*)(ap + kb + 1 * 16 * SH_STRIDE + 16);
    n2f.h[0] = *(const v8h*)(ap + kb + 2 * 16 * SH_STRIDE);
    n2f.h[1] = *(const v8h*)(ap + kb + 2 * 16 * SH_STRIDE + 16);
    n3f.h[0] = *(const v8h*)(ap + kb + 3 * 16 * SH_STRIDE);
    n3f.h[1] = *(const v8h*)(ap + kb + 3 * 16 * SH_STRIDE + 16);

    acc0 = __builtin_amdgcn_wmma_f32_16x16x32_f16(false, a0.v, false, bc.v,
                                                  (short)0, acc0, false, false);
    acc1 = __builtin_amdgcn_wmma_f32_16x16x32_f16(false, a1.v, false, bc.v,
                                                  (short)0, acc1, false, false);
    acc2 = __builtin_amdgcn_wmma_f32_16x16x32_f16(false, a2.v, false, bc.v,
                                                  (short)0, acc2, false, false);
    acc3 = __builtin_amdgcn_wmma_f32_16x16x32_f16(false, a3.v, false, bc.v,
                                                  (short)0, acc3, false, false);
    bc = bn; a0 = n0f; a1 = n1f; a2 = n2f; a3 = n3f;
  }
  acc0 = __builtin_amdgcn_wmma_f32_16x16x32_f16(false, a0.v, false, bc.v,
                                                (short)0, acc0, false, false);
  acc1 = __builtin_amdgcn_wmma_f32_16x16x32_f16(false, a1.v, false, bc.v,
                                                (short)0, acc1, false, false);
  acc2 = __builtin_amdgcn_wmma_f32_16x16x32_f16(false, a2.v, false, bc.v,
                                                (short)0, acc2, false, false);
  acc3 = __builtin_amdgcn_wmma_f32_16x16x32_f16(false, a3.v, false, bc.v,
                                                (short)0, acc3, false, false);

  // ---- epilogue: +b2, store 4 M-tiles x 8 rows ----
  const float bv = b2[n0 + lr];
  float* op = &Out[(size_t)bt * U_DEC * D_OUT];
  #pragma unroll
  for (int r = 0; r < 8; ++r) {
    op[(0 * 16 + r + mofs) * D_OUT + n0 + lr] = acc0[r] + bv;
    op[(1 * 16 + r + mofs) * D_OUT + n0 + lr] = acc1[r] + bv;
    op[(2 * 16 + r + mofs) * D_OUT + n0 + lr] = acc2[r] + bv;
    op[(3 * 16 + r + mofs) * D_OUT + n0 + lr] = acc3[r] + bv;
  }
}

// ---------------------------------------------------------------------------
extern "C" void kernel_launch(void* const* d_in, const int* in_sizes, int n_in,
                              void* d_out, int out_size, void* d_ws, size_t ws_size,
                              hipStream_t stream) {
  const float* enc = (const float*)d_in[0];   // (8,256,512)
  const float* dec = (const float*)d_in[1];   // (8,64,512)
  const float* W1  = (const float*)d_in[2];   // (1024,1024)
  const float* b1  = (const float*)d_in[3];   // (1024,)
  const float* W2  = (const float*)d_in[4];   // (1024,128)
  const float* b2  = (const float*)d_in[5];   // (128,)
  float* out = (float*)d_out;                 // (8,256,64,128)

  float*    enc_proj   = (float*)d_ws;                                 // 8MB
  float*    dec_projb1 = enc_proj + (size_t)BATCH * T_ENC * D_HIDDEN;  // 2MB
  _Float16* W1t        = (_Float16*)(dec_projb1 + (size_t)BATCH * U_DEC * D_HIDDEN); // 2MB
  _Float16* W2t        = W1t + (size_t)1024 * 1024;                    // 256KB

  prep_kernel<<<(1024 * 1024 + 128 * 1024) / 256, 256, 0, stream>>>(W1, W2, W1t, W2t);
  proj_kernel<<<dim3((BATCH * T_ENC) / 64, 8), 256, 0, stream>>>(enc, W1t, nullptr, enc_proj);
  proj_kernel<<<dim3((BATCH * U_DEC) / 64, 8), 256, 0, stream>>>(dec, W1t + 512, b1, dec_projb1);
  joint_kernel<<<BATCH * T_ENC, 256, 0, stream>>>(enc_proj, dec_projb1, W2t, b2, out);
}
